// CogVlmVisionExpertMLP_65927747994049
// MI455X (gfx1250) — matrix operand order repaired and checked by reading, not compiled, not run on hardware
//
#include <hip/hip_runtime.h>
#include <hip/hip_bf16.h>

// CogVLM vision-expert MLP for MI455X (gfx1250, wave32, WMMA).
// B=2, S=2048, H=4096, I=11008. fp32 in/out; bf16 WMMA with fp32 accumulate.
//
// Pipeline (all on `stream`, graph-capture safe, deterministic):
//   1) convert 6 weight matrices + hidden_states fp32 -> bf16 (workspace)
//   2) partition tokens into vision/language index lists (single-wave ballot scan)
//   3) gemm1: h = silu(x@Wg) * (x@Wu) per expert on gathered rows -> bf16 hbuf
//   4) gemm2: out = h@Wd, scattered back to d_out via index lists
//
// GEMM kernels use double-buffered LDS + register prefetch: global loads for
// K-step kb+1 are issued before the WMMA chain of step kb, hiding HBM/L2
// latency at low occupancy (1 barrier per K-step).
//
// Workspace requirement: ~671 MB
//   [0)                bf16 weights: 6 * 4096*11008 halves   = 541,065,216 B
//   [+541065216)       bf16 x:       4096*4096 halves        =  33,554,432 B
//   [+574619648)       bf16 hbuf:    4352*11008 halves       =  95,813,632 B
//   [+670433280)       visIdx[4096], langIdx[4096], cnt[2]   ~      32,776 B

#define B_    2
#define S_    2048
#define H_    4096
#define I_    11008
#define NTOK  (B_ * S_)
#define WELEM ((size_t)H_ * (size_t)I_)   // 45,088,768 elements per weight matrix
#define HROWS 4352                        // padded rows in hbuf (>= 4096 + 2*127)
#define LDSU  (128 * 18)                  // one LDS tile: 128 rows x 18 uints (36 halves)

typedef __attribute__((ext_vector_type(16))) __bf16 v16bf;
typedef __attribute__((ext_vector_type(8)))  float  v8f;

// ---------------------------------------------------------------- helpers ----

__device__ inline unsigned short f2bf(float f) {
  unsigned u = __float_as_uint(f);
  u += 0x7fffu + ((u >> 16) & 1u);      // round-to-nearest-even
  return (unsigned short)(u >> 16);
}

__device__ inline v8f wmma_bf16(v16bf a, v16bf b, v8f c) {
  // D = A(16x32 bf16) * B(32x16 bf16) + C(16x16 f32)
  return __builtin_amdgcn_wmma_f32_16x16x32_bf16(
      /*neg_a=*/false, a, /*neg_b=*/false, b,
      /*c_mod=*/(short)0, c, /*reuse_a=*/false, /*reuse_b=*/false);
}

// LDS tiles are [row][36 halves] == [row][18 uints] (padded to dodge bank conflicts).
// A-matrix 16-bit layout (ISA 7.12.2): lane m<16 holds K={0..7,16..23}, lane m+16
// holds K={8..15,24..31}, 2 K-values per VGPR.
__device__ inline v16bf load_a_frag(const unsigned* As, int mtile, int lane) {
  const int m  = mtile * 16 + (lane & 15);
  const int hi = lane >> 4;
  union { v16bf v; unsigned u[8]; } r;
  const unsigned* row = As + m * 18;
#pragma unroll
  for (int v = 0; v < 8; ++v) {
    int kp = (v < 4 ? v : 8 + (v - 4)) + hi * 4;   // k/2 index
    r.u[v] = row[kp];
  }
  return r.v;
}

// B-matrix 32x16 bf16: lane n<16 holds K={0..7,16..23}, lane n+16 holds K={8..15,24..31}.
__device__ inline v16bf load_b_frag(const unsigned* Bs, int ntile, int lane) {
  const int n  = ntile * 16 + (lane & 15);
  const int kh = lane >> 4;
  union { v16bf v; unsigned u[8]; } r;
  const unsigned* row = Bs + n * 18;
#pragma unroll
  for (int v = 0; v < 8; ++v) r.u[v] = row[(v < 4 ? v : 8 + (v - 4)) + kh * 4];
  return r.v;
}

__device__ inline v8f v8f_zero() {
  v8f z;
#pragma unroll
  for (int j = 0; j < 8; ++j) z[j] = 0.0f;
  return z;
}

// Scatter one 32x128 weight tile held in regs (2 x uint4/thread) into LDS [n][k].
__device__ inline void scatter_b(const uint4 w[2], unsigned* B, int tid) {
  unsigned short* bs = (unsigned short*)B;
#pragma unroll
  for (int i = 0; i < 2; ++i) {
    int q = tid + i * 256;
    int k = q >> 4, n0 = (q & 15) * 8;
    unsigned wv[4] = {w[i].x, w[i].y, w[i].z, w[i].w};
#pragma unroll
    for (int j = 0; j < 4; ++j) {
      bs[(n0 + 2 * j    ) * 36 + k] = (unsigned short)(wv[j] & 0xffffu);
      bs[(n0 + 2 * j + 1) * 36 + k] = (unsigned short)(wv[j] >> 16);
    }
  }
}

// Store one 128x32 activation tile held in regs (2 x uint4/thread) into LDS [m][k].
__device__ inline void store_a(const uint4 a[2], unsigned* A, int tid) {
#pragma unroll
  for (int i = 0; i < 2; ++i) {
    int q = tid + i * 256;
    int row = q >> 2, k0 = (q & 3) * 8;
    unsigned* dst = A + row * 18 + (k0 >> 1);
    dst[0] = a[i].x; dst[1] = a[i].y; dst[2] = a[i].z; dst[3] = a[i].w;
  }
}

// ---------------------------------------------------------------- kernels ----

__global__ __launch_bounds__(256) void cvt_f32_to_bf16(
    const float* __restrict__ src, unsigned short* __restrict__ dst, size_t npairs) {
  size_t i = (size_t)blockIdx.x * blockDim.x + threadIdx.x;
  size_t stride = (size_t)gridDim.x * blockDim.x;
  unsigned* d32 = (unsigned*)dst;
  const float2* s2 = (const float2*)src;
  for (; i < npairs; i += stride) {
    float2 f = s2[i];
    unsigned lo = f2bf(f.x), hi = f2bf(f.y);
    d32[i] = lo | (hi << 16);
  }
}

// Single wave32: stable partition of tokens into vision / language index lists.
__global__ __launch_bounds__(32) void partition_tokens(
    const int* __restrict__ tt, int* __restrict__ visIdx,
    int* __restrict__ langIdx, int* __restrict__ cnt) {
  const int lane = threadIdx.x;
  int vbase = 0, lbase = 0;
  for (int c = 0; c < NTOK; c += 32) {
    int t = c + lane;
    int s = t & (S_ - 1);
    bool vis = (s < S_ - 1) && (tt[t] == 1) && (tt[t + 1] == 1);
    unsigned bal = (unsigned)__ballot(vis);           // wave32: low 32 bits
    int pre = __popc(bal & ((1u << lane) - 1u));
    if (vis) visIdx[vbase + pre] = t;
    else     langIdx[lbase + (lane - pre)] = t;
    int nv = __popc(bal);
    vbase += nv;
    lbase += 32 - nv;
  }
  if (lane == 0) { cnt[0] = vbase; cnt[1] = lbase; }
}

// Stage 1: h = silu(x@Wg) * (x@Wu) on gathered rows of one expert.
// grid = (I/128, expert, rowSplit), block = 256 (8 waves, 4x2 wave tiling).
__global__ __launch_bounds__(256) void gemm1_swiglu(
    const unsigned short* __restrict__ xbf,
    const unsigned short* __restrict__ wbase,
    unsigned short* __restrict__ hbuf,
    const int* __restrict__ visIdx, const int* __restrict__ langIdx,
    const int* __restrict__ cnt) {
  __shared__ unsigned As[2][LDSU];
  __shared__ unsigned Bg[2][LDSU];
  __shared__ unsigned Bu[2][LDSU];

  const int e = blockIdx.y;              // 0 = vision, 1 = language
  const int cntE = cnt[e];
  if (cntE == 0) return;
  const int nvp   = (cnt[0] + 127) & ~127;
  const int hRow0 = (e == 0) ? 0 : nvp;
  const int* idx  = (e == 0) ? visIdx : langIdx;
  const unsigned short* Wg = wbase + (size_t)((e == 0) ? 3 : 0) * WELEM;
  const unsigned short* Wu = Wg + WELEM;

  const int ncol0 = blockIdx.x * 128;
  const int tid = threadIdx.x, lane = tid & 31, wv = tid >> 5;
  const int mw = wv & 3, nw = wv >> 2;   // wave tile: 2 mtiles x 4 ntiles
  const int nK = H_ / 32;

  const int nRB = (cntE + 127) >> 7;
  for (int rb = blockIdx.z; rb < nRB; rb += gridDim.z) {
    int tok[2];
#pragma unroll
    for (int i = 0; i < 2; ++i) {
      int q = tid + i * 256;
      int r = rb * 128 + (q >> 2);
      tok[i] = (r < cntE) ? idx[r] : -1;
    }
    const int ka0 = ((tid & 3) * 8);          // A k-offset for this thread
    const int kw  = ((tid & 255) >> 4);       // B k-row for this thread (i==0)
    const int nb0 = (tid & 15) * 8;           // B n-offset

    uint4 pa[2], pg[2], pu[2];
    // ---- prologue: prefetch kb = 0 ----
#pragma unroll
    for (int i = 0; i < 2; ++i) {
      pa[i] = uint4{0u, 0u, 0u, 0u};
      if (tok[i] >= 0)
        pa[i] = *(const uint4*)(xbf + (size_t)tok[i] * H_ + ka0);
      size_t off = (size_t)(kw + i * 16) * I_ + ncol0 + nb0;
      pg[i] = *(const uint4*)(Wg + off);
      pu[i] = *(const uint4*)(Wu + off);
    }

    v8f aG[2][4], aU[2][4];
#pragma unroll
    for (int mi = 0; mi < 2; ++mi)
#pragma unroll
      for (int ni = 0; ni < 4; ++ni) { aG[mi][ni] = v8f_zero(); aU[mi][ni] = v8f_zero(); }

    for (int kb = 0; kb < nK; ++kb) {
      const int cur = kb & 1;
      // ---- drain prefetched regs into LDS buffer `cur` ----
      store_a(pa, As[cur], tid);
      scatter_b(pg, Bg[cur], tid);
      scatter_b(pu, Bu[cur], tid);
      __syncthreads();                       // compiler drains DScnt here
      // ---- issue global prefetch for kb+1 (overlaps with WMMA below) ----
      if (kb + 1 < nK) {
        const int kg = (kb + 1) * 32;
#pragma unroll
        for (int i = 0; i < 2; ++i) {
          pa[i] = uint4{0u, 0u, 0u, 0u};
          if (tok[i] >= 0)
            pa[i] = *(const uint4*)(xbf + (size_t)tok[i] * H_ + kg + ka0);
          size_t off = (size_t)(kg + kw + i * 16) * I_ + ncol0 + nb0;
          pg[i] = *(const uint4*)(Wg + off);
          pu[i] = *(const uint4*)(Wu + off);
        }
      }
      // ---- WMMA chain on buffer `cur` ----
      v16bf a0 = load_a_frag(As[cur], mw * 2 + 0, lane);
      v16bf a1 = load_a_frag(As[cur], mw * 2 + 1, lane);
#pragma unroll
      for (int ni = 0; ni < 4; ++ni) {
        v16bf bg = load_b_frag(Bg[cur], nw * 4 + ni, lane);
        v16bf bu = load_b_frag(Bu[cur], nw * 4 + ni, lane);
        aG[0][ni] = wmma_bf16(a0, bg, aG[0][ni]);
        aG[1][ni] = wmma_bf16(a1, bg, aG[1][ni]);
        aU[0][ni] = wmma_bf16(a0, bu, aU[0][ni]);
        aU[1][ni] = wmma_bf16(a1, bu, aU[1][ni]);
      }
      // no trailing barrier: next iteration writes the *other* buffer, and its
      // own __syncthreads (with dscnt drain) orders reads-before-overwrite.
    }
    // fused SiLU epilogue -> bf16 intermediate (C layout: VGPR r = M r / r+8).
    // silu via fast v_rcp_f32 (bf16 output makes IEEE division pointless).
    const int hi = lane >> 4, nlo = lane & 15;
#pragma unroll
    for (int mi = 0; mi < 2; ++mi)
#pragma unroll
      for (int ni = 0; ni < 4; ++ni)
#pragma unroll
        for (int el = 0; el < 8; ++el) {
          float g = aG[mi][ni][el];
          float u = aU[mi][ni][el];
          float sig = __builtin_amdgcn_rcpf(1.0f + __expf(-g));  // v_rcp_f32
          float h = g * sig * u;
          int m = (mw * 2 + mi) * 16 + el + hi * 8;
          int n = ncol0 + (nw * 4 + ni) * 16 + nlo;
          size_t row = (size_t)(hRow0 + rb * 128 + m);
          hbuf[row * I_ + n] = f2bf(h);
        }
  }
}

// Stage 2: out = h @ Wd, scattered to d_out rows via index list.
// grid = (H/128, expert, rowSplit), block = 256.
__global__ __launch_bounds__(256) void gemm2_down(
    const unsigned short* __restrict__ hbuf,
    const unsigned short* __restrict__ wbase,
    float* __restrict__ out,
    const int* __restrict__ visIdx, const int* __restrict__ langIdx,
    const int* __restrict__ cnt) {
  __shared__ unsigned As[2][LDSU];
  __shared__ unsigned Bd[2][LDSU];

  const int e = blockIdx.y;
  const int cntE = cnt[e];
  if (cntE == 0) return;
  const int nvp   = (cnt[0] + 127) & ~127;
  const int hRow0 = (e == 0) ? 0 : nvp;
  const int* idx  = (e == 0) ? visIdx : langIdx;
  const unsigned short* Wd = wbase + (size_t)(((e == 0) ? 3 : 0) + 2) * WELEM;

  const int ncol0 = blockIdx.x * 128;
  const int tid = threadIdx.x, lane = tid & 31, wv = tid >> 5;
  const int mw = wv & 3, nw = wv >> 2;
  const int nK = I_ / 32;

  const int nRB = (cntE + 127) >> 7;
  for (int rb = blockIdx.z; rb < nRB; rb += gridDim.z) {
    const int arow = (tid & 255) >> 2;         // A row for this thread (i adds 64)
    const int ka0  = (tid & 3) * 8;
    const int kw   = (tid & 255) >> 4;
    const int nb0  = (tid & 15) * 8;
    const unsigned short* hrow0 =
        hbuf + (size_t)(hRow0 + rb * 128) * I_;

    uint4 pa[2], pw[2];
    // ---- prologue: prefetch kb = 0 ----
#pragma unroll
    for (int i = 0; i < 2; ++i) {
      pa[i] = *(const uint4*)(hrow0 + (size_t)(arow + i * 64) * I_ + ka0);
      size_t off = (size_t)(kw + i * 16) * H_ + ncol0 + nb0;
      pw[i] = *(const uint4*)(Wd + off);
    }

    v8f acc[2][4];
#pragma unroll
    for (int mi = 0; mi < 2; ++mi)
#pragma unroll
      for (int ni = 0; ni < 4; ++ni) acc[mi][ni] = v8f_zero();

    for (int kb = 0; kb < nK; ++kb) {
      const int cur = kb & 1;
      store_a(pa, As[cur], tid);
      scatter_b(pw, Bd[cur], tid);
      __syncthreads();
      if (kb + 1 < nK) {
        const int kg = (kb + 1) * 32;
#pragma unroll
        for (int i = 0; i < 2; ++i) {
          pa[i] = *(const uint4*)(hrow0 + (size_t)(arow + i * 64) * I_ + kg + ka0);
          size_t off = (size_t)(kg + kw + i * 16) * H_ + ncol0 + nb0;
          pw[i] = *(const uint4*)(Wd + off);
        }
      }
      v16bf a0 = load_a_frag(As[cur], mw * 2 + 0, lane);
      v16bf a1 = load_a_frag(As[cur], mw * 2 + 1, lane);
#pragma unroll
      for (int ni = 0; ni < 4; ++ni) {
        v16bf b = load_b_frag(Bd[cur], nw * 4 + ni, lane);
        acc[0][ni] = wmma_bf16(a0, b, acc[0][ni]);
        acc[1][ni] = wmma_bf16(a1, b, acc[1][ni]);
      }
    }
    const int hi = lane >> 4, nlo = lane & 15;
#pragma unroll
    for (int mi = 0; mi < 2; ++mi)
#pragma unroll
      for (int ni = 0; ni < 4; ++ni)
#pragma unroll
        for (int el = 0; el < 8; ++el) {
          int m = (mw * 2 + mi) * 16 + el + hi * 8;
          int r = rb * 128 + m;
          if (r < cntE) {
            int t = idx[r];
            int n = ncol0 + (nw * 4 + ni) * 16 + nlo;
            out[(size_t)t * H_ + n] = acc[mi][ni][el];
          }
        }
  }
}

// ----------------------------------------------------------------- launch ----

extern "C" void kernel_launch(void* const* d_in, const int* in_sizes, int n_in,
                              void* d_out, int out_size, void* d_ws, size_t ws_size,
                              hipStream_t stream) {
  (void)in_sizes; (void)n_in; (void)out_size; (void)ws_size;

  const float* x  = (const float*)d_in[0];
  const int*   tt = (const int*)d_in[1];
  // d_in[2..7]: lang_gate, lang_up, lang_down, vis_gate, vis_up, vis_down (fp32)

  unsigned short* wbf  = (unsigned short*)d_ws;
  unsigned short* xbf  = wbf + 6 * WELEM;
  unsigned short* hbuf = xbf + (size_t)NTOK * H_;
  int* visIdx  = (int*)(hbuf + (size_t)HROWS * I_);
  int* langIdx = visIdx + NTOK;
  int* cnt     = langIdx + NTOK;

  for (int w = 0; w < 6; ++w)
    cvt_f32_to_bf16<<<4096, 256, 0, stream>>>(
        (const float*)d_in[2 + w], wbf + (size_t)w * WELEM, WELEM / 2);
  cvt_f32_to_bf16<<<2048, 256, 0, stream>>>(x, xbf, (size_t)NTOK * H_ / 2);

  partition_tokens<<<1, 32, 0, stream>>>(tt, visIdx, langIdx, cnt);

  gemm1_swiglu<<<dim3(I_ / 128, 2, 2), 256, 0, stream>>>(
      xbf, wbf, hbuf, visIdx, langIdx, cnt);
  gemm2_down<<<dim3(H_ / 128, 2, 8), 256, 0, stream>>>(
      hbuf, wbf, (float*)d_out, visIdx, langIdx, cnt);
}
